// PureGNN_32031866093810
// MI455X (gfx1250) — matrix operand order repaired and checked by reading, not compiled, not run on hardware
//
#include <hip/hip_runtime.h>
#include <cmath>

#define NN 100000
#define NE 1600000
#define F  64
#define BN_EPS 1e-5f

// ---------------- CDNA5 WMMA plumbing (wave32, 16x16x32 bf16) ----------------
typedef __attribute__((ext_vector_type(16))) __bf16 v16bf;
typedef __attribute__((ext_vector_type(8)))  float  v8f;

struct alignas(16) U128 { unsigned int x[4]; };
union Frag16 { U128 u[2]; v16bf v; };

__device__ __forceinline__ v16bf ld_frag(const __bf16* lo, const __bf16* hi) {
  Frag16 f;
  f.u[0] = *reinterpret_cast<const U128*>(lo);   // ds_load_b128
  f.u[1] = *reinterpret_cast<const U128*>(hi);   // ds_load_b128
  return f.v;
}

__device__ __forceinline__ v8f wmma_bf16(v16bf a, v16bf b, v8f c) {
  // (neg_a, A, neg_b, B, c_mod, C, reuse_a, reuse_b)
  return __builtin_amdgcn_wmma_f32_16x16x32_bf16(false, a, false, b, (short)0, c,
                                                 false, false);
}

__device__ __forceinline__ float sigmoidf_(float x) { return 1.f / (1.f + __expf(-x)); }
__device__ __forceinline__ float siluf_(float x)    { return x * sigmoidf_(x); }

// stage one 64x64 f32 weight matrix (row-major [k][n]) into LDS transposed [n][k] bf16
__device__ __forceinline__ void stage_w(__bf16 (*sWm)[F], const float* __restrict__ W,
                                        int tid) {
  #pragma unroll 4
  for (int idx = tid; idx < F * F; idx += 256) {
    int k = idx >> 6, n = idx & 63;
    sWm[n][k] = (__bf16)W[idx];
  }
}

// one 16x64 output tile: acc = A(16x64) @ W(64x64) + bias, streamed to global.
// No guards: caller guarantees the whole tile is in range (wave-uniform).
__device__ __forceinline__ void gemm_tile(const __bf16 (*sWm)[F],
                                          const float* __restrict__ bias,
                                          float* __restrict__ out,
                                          v16bf a0, v16bf a1,
                                          int lane, int hi, int rowtile)
{
  const int kb0 = hi ? 8 : 0;           // B layout: lanes>=16 hold K+8 (K {0..7,16..23}+kb0)
  #pragma unroll
  for (int t = 0; t < 4; ++t) {
    const int col = t * 16 + (lane & 15);
    const float bv = bias[col];
    v8f acc;
    #pragma unroll
    for (int r = 0; r < 8; ++r) acc[r] = bv;
    const __bf16* bp0 = &sWm[col][kb0];
    const __bf16* bp1 = &sWm[col][32 + kb0];
    acc = wmma_bf16(a0, ld_frag(bp0, bp0 + 16), acc);
    acc = wmma_bf16(a1, ld_frag(bp1, bp1 + 16), acc);
    // C layout: vgpr r -> row r + 8*hi, col = lane%16 (within tile t)
    float* op = out + (size_t)(rowtile + 8 * hi) * F + col;
    #pragma unroll
    for (int r = 0; r < 8; ++r) op[r * F] = acc[r];   // global_store_b32, imm offs
  }
}

// ---------------- Kernel 1: fused 4-way node GEMM ----------------------------
// e_src = nf@Wsg+bsg ; e_dst = nf@Wdg+bdg ; Bh = nf@Wdu+bdu ; Ax = nf@Wsu+bsu
__global__ void __launch_bounds__(256)
k_node_gemm(const float* __restrict__ nf,
            const float* __restrict__ W0, const float* __restrict__ W1,
            const float* __restrict__ W2, const float* __restrict__ W3,
            const float* __restrict__ b0, const float* __restrict__ b1,
            const float* __restrict__ b2, const float* __restrict__ b3,
            float* __restrict__ o0, float* __restrict__ o1,
            float* __restrict__ o2, float* __restrict__ o3)
{
  __shared__ __align__(16) __bf16 sW[4][F][F];   // transposed [m][n][k], 32KB
  __shared__ __align__(16) __bf16 sA[8][16][F];  // per-wave A tile,      16KB

  const int tid  = threadIdx.x;
  const int lane = tid & 31, w = tid >> 5;

  stage_w(sW[0], W0, tid);
  stage_w(sW[1], W1, tid);
  stage_w(sW[2], W2, tid);
  stage_w(sW[3], W3, tid);

  // NN % 16 == 0 -> each wave's 16-row tile is fully in range or fully out
  const int rowtile = blockIdx.x * 128 + w * 16;
  const bool live = (rowtile < NN);
  if (live) {
    #pragma unroll
    for (int idx = lane; idx < 256; idx += 32) {
      int r = idx >> 4, ch = idx & 15;
      float4 v = *reinterpret_cast<const float4*>(nf + (size_t)(rowtile + r) * F + ch * 4);
      __bf16* p = &sA[w][r][ch * 4];
      p[0] = (__bf16)v.x; p[1] = (__bf16)v.y; p[2] = (__bf16)v.z; p[3] = (__bf16)v.w;
    }
  }
  __syncthreads();
  if (!live) return;                    // no further barriers below

  // A fragments per ISA layout: lane<16 -> K {0..7,16..23}+k0, lane>=16 -> +8
  const int hi = (lane >> 4) & 1;
  const __bf16* arow = &sA[w][lane & 15][0];
  const int aoff = hi ? 8 : 0;
  v16bf a0 = ld_frag(arow + aoff,      arow + aoff + 16);       // K 0..31
  v16bf a1 = ld_frag(arow + 32 + aoff, arow + 32 + aoff + 16);  // K 32..63

  gemm_tile(sW[0], b0, o0, a0, a1, lane, hi, rowtile);
  gemm_tile(sW[1], b1, o1, a0, a1, lane, hi, rowtile);
  gemm_tile(sW[2], b2, o2, a0, a1, lane, hi, rowtile);
  gemm_tile(sW[3], b3, o3, a0, a1, lane, hi, rowtile);
}

// ---------------- Kernel 2/6: edge GEMM + gather (+scatter or +output) -------
template<bool PASS2>
__global__ void __launch_bounds__(256)
k_edge(const float* __restrict__ ef, const int* __restrict__ src,
       const int* __restrict__ dst,
       const float* __restrict__ Wg, const float* __restrict__ bg,
       const float* __restrict__ esrc, const float* __restrict__ edst,
       const float* __restrict__ Bh, float* __restrict__ ssh,
       float* __restrict__ ss, float* __restrict__ estat,
       const float* __restrict__ escale, const float* __restrict__ eshift,
       float* __restrict__ yout)
{
  __shared__ __align__(16) __bf16 sW[F][F];       //  8KB
  __shared__ __align__(16) __bf16 sA[8][16][F];   // 16KB
  __shared__ __align__(16) float  sM[8][16][F];   // 32KB
  __shared__ float sStat[128];

  const int tid  = threadIdx.x;
  const int lane = tid & 31, w = tid >> 5;
  const int etile = blockIdx.x * 128 + w * 16;    // NE % 128 == 0, no guards

  stage_w(sW, Wg, tid);
  #pragma unroll
  for (int idx = lane; idx < 256; idx += 32) {
    int r = idx >> 4, ch = idx & 15;
    float4 v = *reinterpret_cast<const float4*>(ef + (size_t)(etile + r) * F + ch * 4);
    __bf16* p = &sA[w][r][ch * 4];
    p[0] = (__bf16)v.x; p[1] = (__bf16)v.y; p[2] = (__bf16)v.z; p[3] = (__bf16)v.w;
  }
  if (!PASS2 && tid < 128) sStat[tid] = 0.f;
  __syncthreads();

  const int hi = (lane >> 4) & 1;
  const __bf16* arow = &sA[w][lane & 15][0];
  const int aoff = hi ? 8 : 0;
  v16bf a0 = ld_frag(arow + aoff,      arow + aoff + 16);
  v16bf a1 = ld_frag(arow + 32 + aoff, arow + 32 + aoff + 16);

  const int kb0 = hi ? 8 : 0;
  #pragma unroll
  for (int t = 0; t < 4; ++t) {
    const int col = t * 16 + (lane & 15);
    const float bv = bg[col];
    v8f acc;
    #pragma unroll
    for (int r = 0; r < 8; ++r) acc[r] = bv;
    const __bf16* bp0 = &sW[col][kb0];
    const __bf16* bp1 = &sW[col][32 + kb0];
    acc = wmma_bf16(a0, ld_frag(bp0, bp0 + 16), acc);
    acc = wmma_bf16(a1, ld_frag(bp1, bp1 + 16), acc);
    float* mp = &sM[w][8 * hi][col];
    #pragma unroll
    for (int r = 0; r < 8; ++r) mp[r * F] = acc[r];   // ds_store_b32, imm offs
  }
  __syncthreads();

  // elementwise view: lane owns column chunk q4, lane-half interleaves edges
  const int q4 = (lane & 15) * 4;
  float ps[4] = {0, 0, 0, 0}, pq[4] = {0, 0, 0, 0};
  #pragma unroll
  for (int rep = 0; rep < 8; ++rep) {
    const int el = hi + 2 * rep;
    const int e  = etile + el;
    const int s  = src[e], d = dst[e];
    const float* mrow = &sM[w][el][q4];
    const float4 gs4 = *reinterpret_cast<const float4*>(esrc + (size_t)s * F + q4);
    const float4 gd4 = *reinterpret_cast<const float4*>(edst + (size_t)d * F + q4);
    float mv[4];
    mv[0] = mrow[0] + gs4.x + gd4.x;
    mv[1] = mrow[1] + gs4.y + gd4.y;
    mv[2] = mrow[2] + gs4.z + gd4.z;
    mv[3] = mrow[3] + gs4.w + gd4.w;
    if (!PASS2) {
      const float4 bh4 = *reinterpret_cast<const float4*>(Bh + (size_t)s * F + q4);
      const float bh[4] = {bh4.x, bh4.y, bh4.z, bh4.w};
      float* pssh = ssh + (size_t)d * F + q4;
      float* pss  = ss  + (size_t)d * F + q4;
      #pragma unroll
      for (int j = 0; j < 4; ++j) {
        float sg = sigmoidf_(mv[j]);
        atomicAdd(pssh + j, sg * bh[j]);
        atomicAdd(pss + j, sg);
        ps[j] += mv[j];
        pq[j] += mv[j] * mv[j];
      }
    } else {
      const float4 ef4 = *reinterpret_cast<const float4*>(ef + (size_t)e * F + q4);
      const float efv[4] = {ef4.x, ef4.y, ef4.z, ef4.w};
      float4 r4;
      float rv[4];
      #pragma unroll
      for (int j = 0; j < 4; ++j) {
        float tt = mv[j] * escale[q4 + j] + eshift[q4 + j];
        rv[j] = efv[j] + siluf_(tt);
      }
      r4.x = rv[0]; r4.y = rv[1]; r4.z = rv[2]; r4.w = rv[3];
      *reinterpret_cast<float4*>(yout + (size_t)e * F + q4) = r4;
    }
  }
  if (!PASS2) {
    #pragma unroll
    for (int j = 0; j < 4; ++j) {
      atomicAdd(&sStat[q4 + j],      ps[j]);
      atomicAdd(&sStat[64 + q4 + j], pq[j]);
    }
    __syncthreads();
    if (tid < 128) atomicAdd(&estat[tid], sStat[tid]);
  }
}

// ---------------- Kernel 3: node aggregation + node BN stats -----------------
__global__ void __launch_bounds__(256)
k_node_agg(const float* __restrict__ Ax, const float* __restrict__ ssh,
           const float* __restrict__ ss, float* __restrict__ xpre,
           float* __restrict__ nstat)
{
  __shared__ float sStat[128];
  const int tid = threadIdx.x;
  if (tid < 128) sStat[tid] = 0.f;
  __syncthreads();
  const int i  = blockIdx.x * 256 + tid;     // exactly NN*16 tasks
  const int n  = i >> 4, q4 = (i & 15) * 4;
  const size_t base = (size_t)n * F + q4;
  const float4 num = *reinterpret_cast<const float4*>(ssh + base);
  const float4 den = *reinterpret_cast<const float4*>(ss + base);
  const float4 ax  = *reinterpret_cast<const float4*>(Ax + base);
  float x[4];
  x[0] = ax.x + num.x / (den.x + 1e-6f);
  x[1] = ax.y + num.y / (den.y + 1e-6f);
  x[2] = ax.z + num.z / (den.z + 1e-6f);
  x[3] = ax.w + num.w / (den.w + 1e-6f);
  float4 xo; xo.x = x[0]; xo.y = x[1]; xo.z = x[2]; xo.w = x[3];
  *reinterpret_cast<float4*>(xpre + base) = xo;
  #pragma unroll
  for (int j = 0; j < 4; ++j) {
    atomicAdd(&sStat[q4 + j],      x[j]);
    atomicAdd(&sStat[64 + q4 + j], x[j] * x[j]);
  }
  __syncthreads();
  if (tid < 128) atomicAdd(&nstat[tid], sStat[tid]);
}

// ---------------- Kernel 4: fold BN stats into scale/shift -------------------
__global__ void k_finalize(const float* __restrict__ estat, const float* __restrict__ nstat,
                           const float* __restrict__ ge, const float* __restrict__ be,
                           const float* __restrict__ gn, const float* __restrict__ bnb,
                           float* __restrict__ escale, float* __restrict__ eshift,
                           float* __restrict__ nscale, float* __restrict__ nshift)
{
  int t = threadIdx.x;  // 64
  {
    float mean = estat[t] / (float)NE;
    float var  = estat[64 + t] / (float)NE - mean * mean;
    float sc   = ge[t] * rsqrtf(var + BN_EPS);
    escale[t] = sc; eshift[t] = be[t] - mean * sc;
  }
  {
    float mean = nstat[t] / (float)NN;
    float var  = nstat[64 + t] / (float)NN - mean * mean;
    float sc   = gn[t] * rsqrtf(var + BN_EPS);
    nscale[t] = sc; nshift[t] = bnb[t] - mean * sc;
  }
}

// ---------------- Kernel 5: node output (BN + SiLU + residual) ---------------
__global__ void __launch_bounds__(256)
k_node_out(const float* __restrict__ nf, const float* __restrict__ xpre,
           const float* __restrict__ nscale, const float* __restrict__ nshift,
           float* __restrict__ xout)
{
  const int i  = blockIdx.x * 256 + threadIdx.x;
  const int n  = i >> 4, q4 = (i & 15) * 4;
  const size_t base = (size_t)n * F + q4;
  const float4 xp = *reinterpret_cast<const float4*>(xpre + base);
  const float4 nv = *reinterpret_cast<const float4*>(nf + base);
  const float xv[4] = {xp.x, xp.y, xp.z, xp.w};
  const float nfv[4] = {nv.x, nv.y, nv.z, nv.w};
  float rv[4];
  #pragma unroll
  for (int j = 0; j < 4; ++j) {
    float tt = xv[j] * nscale[q4 + j] + nshift[q4 + j];
    rv[j] = nfv[j] + siluf_(tt);
  }
  float4 r4; r4.x = rv[0]; r4.y = rv[1]; r4.z = rv[2]; r4.w = rv[3];
  *reinterpret_cast<float4*>(xout + base) = r4;
}

// ---------------- Host launcher ----------------------------------------------
extern "C" void kernel_launch(void* const* d_in, const int* in_sizes, int n_in,
                              void* d_out, int out_size, void* d_ws, size_t ws_size,
                              hipStream_t stream)
{
  const float* nf  = (const float*)d_in[0];
  const float* ef  = (const float*)d_in[1];
  const int*   src = (const int*)d_in[2];
  const int*   dst = (const int*)d_in[3];
  const float* Wsg = (const float*)d_in[4];
  const float* bsg = (const float*)d_in[5];
  const float* Wdg = (const float*)d_in[6];
  const float* bdg = (const float*)d_in[7];
  const float* Weg = (const float*)d_in[8];
  const float* beg = (const float*)d_in[9];
  const float* Wsu = (const float*)d_in[10];
  const float* bsu = (const float*)d_in[11];
  const float* Wdu = (const float*)d_in[12];
  const float* bdu = (const float*)d_in[13];
  const float* gn  = (const float*)d_in[14];
  const float* bnb = (const float*)d_in[15];
  const float* ge  = (const float*)d_in[16];
  const float* be  = (const float*)d_in[17];

  const size_t NF = (size_t)NN * F;
  float* ws     = (float*)d_ws;
  float* e_src  = ws;
  float* e_dst  = ws + 1 * NF;
  float* Bh     = ws + 2 * NF;
  float* Ax     = ws + 3 * NF;
  float* ssh    = ws + 4 * NF;
  float* ss     = ws + 5 * NF;
  float* estat  = ws + 6 * NF;      // 128 floats
  float* nstat  = estat + 128;      // 128 floats
  float* escale = nstat + 128;      // 64
  float* eshift = escale + 64;      // 64
  float* nscale = eshift + 64;      // 64
  float* nshift = nscale + 64;      // 64
  float* xpre   = nshift + 64;      // NF floats

  // zero scatter accumulators + BN stat accumulators (graph-capture safe)
  hipMemsetAsync(ssh, 0, (2 * NF + 256) * sizeof(float), stream);

  k_node_gemm<<<(NN + 127) / 128, 256, 0, stream>>>(
      nf, Wsg, Wdg, Wdu, Wsu, bsg, bdg, bdu, bsu, e_src, e_dst, Bh, Ax);

  k_edge<false><<<NE / 128, 256, 0, stream>>>(
      ef, src, dst, Weg, beg, e_src, e_dst, Bh, ssh, ss, estat,
      nullptr, nullptr, nullptr);

  k_node_agg<<<(NN * 16) / 256, 256, 0, stream>>>(Ax, ssh, ss, xpre, nstat);

  k_finalize<<<1, 64, 0, stream>>>(estat, nstat, ge, be, gn, bnb,
                                   escale, eshift, nscale, nshift);

  k_node_out<<<(NN * 16) / 256, 256, 0, stream>>>(nf, xpre, nscale, nshift,
                                                  (float*)d_out);

  k_edge<true><<<NE / 128, 256, 0, stream>>>(
      ef, src, dst, Weg, beg, e_src, e_dst, nullptr, nullptr, nullptr, nullptr,
      escale, eshift, (float*)d_out + NF);
}
